// GatEncoder_raw_gru_60971355734174
// MI455X (gfx1250) — matrix operand order repaired and checked by reading, not compiled, not run on hardware
//
#include <hip/hip_runtime.h>
#include <hip/hip_bf16.h>
#include <math.h>

// ---------------------------------------------------------------------------
// GAT encoder + GRU readout for MI455X (gfx1250, wave32, WMMA).
// Node GEMMs: v_wmma_f32_16x16x32_f16, A staged via global_load_async_to_lds,
// weights pre-transposed/padded to K-major f16 so all inner loops are
// guard-free vector loads.  Edge softmax/aggregation via coalesced
// wave-per-edge float4 + f32 global atomics.
// ---------------------------------------------------------------------------

typedef __attribute__((ext_vector_type(16))) _Float16 v16h;
typedef __attribute__((ext_vector_type(8)))  float    v8f;
typedef __attribute__((address_space(3)))    float    lds_f32;

#define N_GRAPHS  256
#define F0        74
#define KP0       96          // F0 padded to multiple of 32
#define HDIM      128
#define HEADS     4
#define CONVS     3
#define NCOL      128         // all GEMMs here have N == 128

// ---------------------------------------------------------------- utilities
__global__ void fill_u32_kernel(unsigned* __restrict__ p, unsigned v, int n) {
  int i = blockIdx.x * blockDim.x + threadIdx.x;
  if (i < n) p[i] = v;
}

// dst[M][Kpad] = src[M][K] zero-padded in K
__global__ void pad_rows_kernel(const float* __restrict__ src, float* __restrict__ dst,
                                int M, int K, int Kpad) {
  int i = blockIdx.x * blockDim.x + threadIdx.x;
  if (i >= M * Kpad) return;
  int r = i / Kpad, k = i - r * Kpad;
  dst[i] = (k < K) ? src[(size_t)r * K + k] : 0.0f;
}

// batched: dst[b][n][Kpad](f16) = src[b][k][n](f32), zero-padded in K
__global__ void wT_f16_kernel(const float* __restrict__ src, _Float16* __restrict__ dst,
                              int K, int Kpad, int count) {
  int i = blockIdx.x * blockDim.x + threadIdx.x;
  int per = NCOL * Kpad;
  if (i >= count * per) return;
  int b = i / per, rem = i - b * per;
  int n = rem / Kpad, k = rem - n * Kpad;
  float v = (k < K) ? src[(size_t)b * K * NCOL + (size_t)k * NCOL + n] : 0.0f;
  dst[i] = (_Float16)v;
}

__device__ __forceinline__ unsigned ord_f32(float f) {
  unsigned u = __float_as_uint(f);
  return (u & 0x80000000u) ? ~u : (u | 0x80000000u);
}
__device__ __forceinline__ float unord_f32(unsigned o) {
  return (o & 0x80000000u) ? __uint_as_float(o & 0x7FFFFFFFu) : __uint_as_float(~o);
}
__device__ __forceinline__ float sigmoidf_(float x) { return 1.0f / (1.0f + expf(-x)); }

// ------------------------------------------------------------- WMMA GEMM
// C[M,128] = A[M,K](f32) * B(f16, pre-transposed [128][K]) (+bias[128]).
// K % 32 == 0 (padded).  Block = 128 threads = 4 waves; block computes a
// 64 x 128 output tile; each wave owns a 16 x 128 strip (8 wmma tiles).
// A tile (64x32 f32) staged into LDS with global_load_async_to_lds_b128.
__global__ void gemm_wmma_kernel(const float* __restrict__ A,
                                 const _Float16* __restrict__ Bt,
                                 float* __restrict__ C,
                                 const float* __restrict__ bias,
                                 int M, int K, int ldC) {
  __shared__ __align__(16) float As[64][36];     // 32 cols + pad, rows 16B aligned
  const int wave = threadIdx.x >> 5;
  const int lane = threadIdx.x & 31;
  const int l16  = lane & 15;
  const int hi   = lane >> 4;                    // 0/1 lane half
  const int blkM = blockIdx.x * 64;
  const bool full = (blkM + 64 <= M);
  const int srow = threadIdx.x >> 1;             // staging row 0..63
  const int sseg = (threadIdx.x & 1) * 16;       // staging col segment
  const int gr   = blkM + srow;

  v8f acc[8];
  #pragma unroll
  for (int ct = 0; ct < 8; ++ct) acc[ct] = (v8f){};

  for (int k0 = 0; k0 < K; k0 += 32) {
    if (full) {
      // async DMA: 4 x 16B per thread, global -> LDS (ASYNCcnt tracked)
      unsigned lofs = (unsigned)(size_t)(lds_f32*)&As[srow][sseg];
      unsigned long long ga =
          (unsigned long long)(size_t)(A + (size_t)gr * K + k0 + sseg);
      asm volatile("global_load_async_to_lds_b128 %0, %1, off"
                   :: "v"(lofs), "v"(ga) : "memory");
      asm volatile("global_load_async_to_lds_b128 %0, %1, off offset:16"
                   :: "v"(lofs), "v"(ga) : "memory");
      asm volatile("global_load_async_to_lds_b128 %0, %1, off offset:32"
                   :: "v"(lofs), "v"(ga) : "memory");
      asm volatile("global_load_async_to_lds_b128 %0, %1, off offset:48"
                   :: "v"(lofs), "v"(ga) : "memory");
    } else {
      #pragma unroll
      for (int c = 0; c < 16; ++c)
        As[srow][sseg + c] = (gr < M) ? A[(size_t)gr * K + k0 + sseg + c] : 0.0f;
    }
    asm volatile("s_wait_asynccnt 0x0" ::: "memory");
    __syncthreads();

    // A fragment (contiguous k order, same as B): lanes 0-15 K0-15, lanes 16-31 K16-31
    v16h af;
    const int arow = wave * 16 + l16;
    const int kb   = hi * 16;
    #pragma unroll
    for (int c = 0; c < 16; ++c) {
      af[c]     = (_Float16)As[arow][kb + c];
    }
    // 8 column tiles: B fragment is 16 contiguous K-major halves (32B load)
    #pragma unroll
    for (int ct = 0; ct < 8; ++ct) {
      const int col = ct * 16 + l16;
      v16h bf = *(const v16h*)(Bt + (size_t)col * K + k0 + hi * 16);
      acc[ct] = __builtin_amdgcn_wmma_f32_16x16x32_f16(false, af, false, bf,
                                                       (short)0, acc[ct], false, false);
    }
    __syncthreads();
  }

  // epilogue: C/D layout — VGPR r: lanes 0-15 -> M=r, lanes 16-31 -> M=8+r
  #pragma unroll
  for (int ct = 0; ct < 8; ++ct) {
    int col = ct * 16 + l16;
    float bv = bias ? bias[col] : 0.0f;
    #pragma unroll
    for (int r = 0; r < 8; ++r) {
      int orow = blkM + wave * 16 + (hi ? (8 + r) : r);
      if (orow < M) C[(size_t)orow * ldC + col] = acc[ct][r] + bv;
    }
  }
}

// -------------------------------------------------- attention scalar scores
__global__ void att_scores_kernel(const float* __restrict__ z,
                                  const float* __restrict__ att,
                                  float* __restrict__ ssrc, float* __restrict__ sdst, int n) {
  int i = blockIdx.x * blockDim.x + threadIdx.x;
  if (i >= n) return;
  const float* zr = z + (size_t)i * HDIM;
  float s0 = 0.f, s1 = 0.f;
  #pragma unroll 4
  for (int f = 0; f < HDIM; ++f) {
    float v = zr[f];
    s0 += v * att[f];
    s1 += v * att[HDIM + f];
  }
  ssrc[i] = s0;
  sdst[i] = s1;
}

// ------------------------------------------------------------- edge kernels
__global__ void edge_score_kernel(const float* __restrict__ ssrc, const float* __restrict__ sdst,
                                  const int* __restrict__ src, const int* __restrict__ dst,
                                  float* __restrict__ e, unsigned* __restrict__ m, int E) {
  int i = blockIdx.x * blockDim.x + threadIdx.x;
  if (i >= E) return;
  float v = ssrc[src[i]] + sdst[dst[i]];
  v = (v > 0.f) ? v : 0.01f * v;            // jax.nn.leaky_relu default slope
  e[i] = v;
  atomicMax(&m[dst[i]], ord_f32(v));
}

__global__ void edge_expsum_kernel(float* __restrict__ e,
                                   const unsigned* __restrict__ m, float* __restrict__ s,
                                   const int* __restrict__ dst, int E) {
  int i = blockIdx.x * blockDim.x + threadIdx.x;
  if (i >= E) return;
  int d = dst[i];
  float ex = expf(e[i] - unord_f32(m[d]));
  e[i] = ex;
  atomicAdd(&s[d], ex);
}

// one wave (32 lanes) per edge, float4 per lane -> coalesced 512B row move
__global__ void edge_agg_kernel(const float* __restrict__ e, const float* __restrict__ s,
                                const float* __restrict__ z,
                                const int* __restrict__ src, const int* __restrict__ dst,
                                float* __restrict__ agg, int E) {
  long long t = (long long)blockIdx.x * blockDim.x + threadIdx.x;
  int edge = (int)(t >> 5);
  int lane = (int)(t & 31);
  if (edge >= E) return;
  int sN = src[edge], dN = dst[edge];
  float a = e[edge] / s[dN];
  const float4 zv = ((const float4*)(z + (size_t)sN * HDIM))[lane];
  float* ar = agg + (size_t)dN * HDIM + lane * 4;
  atomicAdd(ar + 0, a * zv.x);
  atomicAdd(ar + 1, a * zv.y);
  atomicAdd(ar + 2, a * zv.z);
  atomicAdd(ar + 3, a * zv.w);
}

// ------------------------------------------------------- BatchNorm (+ReLU)
__global__ void bn_stats_kernel(const float* __restrict__ agg,
                                float* __restrict__ sum, float* __restrict__ sumsq, int n) {
  int f = threadIdx.x;                       // 128 threads = feature
  int rows = (n + gridDim.x - 1) / gridDim.x;
  int r0 = blockIdx.x * rows;
  int r1 = r0 + rows; if (r1 > n) r1 = n;
  float s = 0.f, q = 0.f;
  for (int r = r0; r < r1; ++r) {
    float v = agg[(size_t)r * HDIM + f];
    v = v > 0.f ? v : 0.f;                   // fused ReLU
    s += v; q += v * v;
  }
  atomicAdd(&sum[f], s);
  atomicAdd(&sumsq[f], q);
}

__global__ void bn_apply_kernel(const float* __restrict__ agg,
                                const float* __restrict__ sum, const float* __restrict__ sumsq,
                                const float* __restrict__ gamma, const float* __restrict__ beta,
                                float* __restrict__ concat, int n, int head) {
  int i = blockIdx.x * blockDim.x + threadIdx.x;
  if (i >= n * HDIM) return;
  int r = i >> 7, f = i & 127;
  float inv_n = 1.0f / (float)n;
  float mu  = sum[f] * inv_n;
  float var = sumsq[f] * inv_n - mu * mu;
  float v = agg[i]; v = v > 0.f ? v : 0.f;
  float o = gamma[f] * (v - mu) * rsqrtf(var + 1e-5f) + beta[f];
  concat[(size_t)r * (HEADS * HDIM) + head * HDIM + f] = o;
}

// --------------------------------------------------------- graph readouts
__global__ void graph_count_kernel(const int* __restrict__ gids, float* __restrict__ gcount, int n) {
  int i = blockIdx.x * blockDim.x + threadIdx.x;
  if (i < n) atomicAdd(&gcount[gids[i]], 1.0f);
}

__global__ void graph_accum_kernel(const float* __restrict__ h, const int* __restrict__ gids,
                                   float* __restrict__ gsum, int n, int F) {
  int i = blockIdx.x * blockDim.x + threadIdx.x;
  if (i >= n * F) return;
  int r = i / F, f = i - r * F;
  atomicAdd(&gsum[(size_t)gids[r] * F + f], h[i]);
}

__global__ void graph_mean_kernel(const float* __restrict__ gsum, const float* __restrict__ gcount,
                                  float* __restrict__ out, int F) {
  int i = blockIdx.x * blockDim.x + threadIdx.x;
  if (i >= N_GRAPHS * F) return;
  int g = i / F;
  out[i] = gsum[i] / fmaxf(gcount[g], 1.0f);
}

// ----------------------------------------------------------------- GRU
struct GruDir { const float *Wih, *Whh, *bih, *bhh; };
struct GruAll { GruDir d[4]; };               // l0f, l0b, l1f, l1b

__global__ void gru_kernel(const float* __restrict__ x0, GruAll P, float* __restrict__ out) {
  __shared__ float xin[4][256];
  __shared__ float ynew[4][256];
  __shared__ float xw[4][3 * HDIM];
  __shared__ float g[3 * HDIM];
  __shared__ float h[HDIM];
  __shared__ float fin[HDIM];
  const int b = blockIdx.x;
  const int t = threadIdx.x;                  // 128 threads = hidden feature

  #pragma unroll
  for (int tt = 0; tt < 4; ++tt)
    xin[tt][t] = x0[(size_t)tt * N_GRAPHS * HDIM + (size_t)b * HDIM + t];
  fin[t] = 0.f;
  __syncthreads();

  for (int layer = 0; layer < 2; ++layer) {
    const int in_dim = (layer == 0) ? HDIM : 2 * HDIM;
    for (int dir = 0; dir < 2; ++dir) {
      const GruDir pp = P.d[layer * 2 + dir];
      for (int tt = 0; tt < 4; ++tt)
        #pragma unroll
        for (int rr = 0; rr < 3; ++rr) {
          int r = t + rr * HDIM;
          float s = pp.bih[r];
          for (int k = 0; k < in_dim; ++k) s += pp.Wih[(size_t)r * in_dim + k] * xin[tt][k];
          xw[tt][r] = s;
        }
      h[t] = 0.f;
      __syncthreads();
      for (int step = 0; step < 4; ++step) {
        int tt = (dir == 0) ? step : (3 - step);
        #pragma unroll
        for (int rr = 0; rr < 3; ++rr) {
          int r = t + rr * HDIM;
          float s = pp.bhh[r];
          for (int k = 0; k < HDIM; ++k) s += pp.Whh[(size_t)r * HDIM + k] * h[k];
          g[r] = s;
        }
        float rg = sigmoidf_(xw[tt][t] + g[t]);
        float zg = sigmoidf_(xw[tt][HDIM + t] + g[HDIM + t]);
        float cg = tanhf(xw[tt][2 * HDIM + t] + rg * g[2 * HDIM + t]);
        float hv = (1.f - zg) * cg + zg * h[t];
        __syncthreads();                      // all h reads done before update
        h[t] = hv;
        if (layer == 0) ynew[tt][dir * HDIM + t] = hv;
        __syncthreads();
      }
      fin[t] += h[t];
      __syncthreads();
    }
    if (layer == 0) {
      #pragma unroll
      for (int tt = 0; tt < 4; ++tt) {
        xin[tt][t] = ynew[tt][t];
        xin[tt][HDIM + t] = ynew[tt][HDIM + t];
      }
      __syncthreads();
    }
  }
  out[(size_t)b * HDIM + t] = fin[t] * 0.25f;
}

// ------------------------------------------------------------------ driver
static inline size_t alignUp(size_t x, size_t a) { return (x + a - 1) / a * a; }
static inline int cdiv(int a, int b) { return (a + b - 1) / b; }

extern "C" void kernel_launch(void* const* d_in, const int* in_sizes, int n_in,
                              void* d_out, int out_size, void* d_ws, size_t ws_size,
                              hipStream_t stream) {
  (void)n_in; (void)out_size; (void)ws_size;
  // input order: h, gat_W0, gat_Wr, gat_att, bn_gamma, bn_beta, out_W, out_b,
  //              proj_W, proj_b, 16x gru params, src, dst, graph_ids
  const float* h0       = (const float*)d_in[0];
  const float* gat_W0   = (const float*)d_in[1];
  const float* gat_Wr   = (const float*)d_in[2];
  const float* gat_att  = (const float*)d_in[3];
  const float* bn_gamma = (const float*)d_in[4];
  const float* bn_beta  = (const float*)d_in[5];
  const float* out_W    = (const float*)d_in[6];
  const float* out_b    = (const float*)d_in[7];
  const float* proj_W   = (const float*)d_in[8];
  const float* proj_b   = (const float*)d_in[9];
  const int GRU0 = 10;   // Wih_0f, Whh_0f, bih_0f, bhh_0f, Wih_0b, ... bhh_1b
  const int* srcI = (const int*)d_in[26];
  const int* dstI = (const int*)d_in[27];
  const int* gids = (const int*)d_in[28];

  const int Nn = in_sizes[0] / F0;
  const int E  = in_sizes[26];

  // ---- workspace carving
  char* ws = (char*)d_ws;
  size_t off = 0;
  auto carve = [&](size_t bytes) { char* p = ws + off; off = alignUp(off + bytes, 256); return p; };
  _Float16* W0h    = (_Float16*)carve((size_t)HEADS * NCOL * KP0 * 2);          // [head][n][KP0]
  _Float16* Wrh    = (_Float16*)carve((size_t)(CONVS - 1) * HEADS * NCOL * HDIM * 2);
  _Float16* outWh  = (_Float16*)carve((size_t)CONVS * NCOL * (HEADS * HDIM) * 2);
  _Float16* projWh = (_Float16*)carve((size_t)NCOL * KP0 * 2);
  float*    h0pad  = (float*)carve((size_t)Nn * KP0 * 4);
  float*    h_cur  = (float*)carve((size_t)Nn * HDIM * 4);
  float*    zbuf   = (float*)carve((size_t)Nn * HDIM * 4);
  float*    agg    = (float*)carve((size_t)Nn * HDIM * 4);
  float*    concat = (float*)carve((size_t)Nn * HEADS * HDIM * 4);
  float*    ssrc   = (float*)carve((size_t)Nn * 4);
  float*    sdst   = (float*)carve((size_t)Nn * 4);
  float*    ebuf   = (float*)carve((size_t)E * 4);
  unsigned* mbuf   = (unsigned*)carve((size_t)Nn * 4);
  float*    sbuf   = (float*)carve((size_t)Nn * 4);
  float*    bnsum  = (float*)carve(256 * 4);          // [0:128] sum, [128:256] sumsq
  float*    gsum   = (float*)carve((size_t)N_GRAPHS * HDIM * 4);
  float*    gcount = (float*)carve((size_t)N_GRAPHS * 4);
  float*    mean74 = (float*)carve((size_t)N_GRAPHS * F0 * 4);
  float*    mean74p= (float*)carve((size_t)N_GRAPHS * KP0 * 4);
  float*    gru_x  = (float*)carve((size_t)4 * N_GRAPHS * HDIM * 4);

  auto fill0 = [&](void* p, int n) {
    fill_u32_kernel<<<cdiv(n, 256), 256, 0, stream>>>((unsigned*)p, 0u, n);
  };
  auto gemm = [&](const float* A, const _Float16* Bt, float* C, const float* bias,
                  int M, int K, int ldC) {
    gemm_wmma_kernel<<<cdiv(M, 64), 128, 0, stream>>>(A, Bt, C, bias, M, K, ldC);
  };

  // ---- one-time weight transpose+pad (fp32 [K,N] -> f16 [N,Kpad])
  wT_f16_kernel<<<cdiv(HEADS * NCOL * KP0, 256), 256, 0, stream>>>(gat_W0, W0h, F0, KP0, HEADS);
  wT_f16_kernel<<<cdiv((CONVS-1)*HEADS * NCOL * HDIM, 256), 256, 0, stream>>>(
      gat_Wr, Wrh, HDIM, HDIM, (CONVS - 1) * HEADS);
  wT_f16_kernel<<<cdiv(CONVS * NCOL * HEADS * HDIM, 256), 256, 0, stream>>>(
      out_W, outWh, HEADS * HDIM, HEADS * HDIM, CONVS);
  wT_f16_kernel<<<cdiv(NCOL * KP0, 256), 256, 0, stream>>>(proj_W, projWh, F0, KP0, 1);
  // pad node features once: [Nn,74] -> [Nn,96]
  pad_rows_kernel<<<cdiv(Nn * KP0, 256), 256, 0, stream>>>(h0, h0pad, Nn, F0, KP0);

  // ---- graph counts + initial readout: mean(h0) [G,74] -> proj -> gru_x[0]
  fill0(gcount, N_GRAPHS);
  graph_count_kernel<<<cdiv(Nn, 256), 256, 0, stream>>>(gids, gcount, Nn);
  fill0(gsum, N_GRAPHS * F0);
  graph_accum_kernel<<<cdiv(Nn * F0, 256), 256, 0, stream>>>(h0, gids, gsum, Nn, F0);
  graph_mean_kernel<<<cdiv(N_GRAPHS * F0, 256), 256, 0, stream>>>(gsum, gcount, mean74, F0);
  pad_rows_kernel<<<cdiv(N_GRAPHS * KP0, 256), 256, 0, stream>>>(mean74, mean74p, N_GRAPHS, F0, KP0);
  gemm(mean74p, projWh, gru_x, proj_b, N_GRAPHS, KP0, HDIM);

  // ---- GAT convs
  for (int j = 0; j < CONVS; ++j) {
    const float* Ain = (j == 0) ? h0pad : h_cur;
    const int Kj = (j == 0) ? KP0 : HDIM;
    for (int i = 0; i < HEADS; ++i) {
      const _Float16* Wh = (j == 0) ? (W0h + (size_t)i * NCOL * KP0)
                                    : (Wrh + (size_t)((j - 1) * HEADS + i) * NCOL * HDIM);
      gemm(Ain, Wh, zbuf, nullptr, Nn, Kj, HDIM);                 // z = A @ W (WMMA)
      att_scores_kernel<<<cdiv(Nn, 256), 256, 0, stream>>>(
          zbuf, gat_att + (size_t)(j * HEADS + i) * 2 * HDIM, ssrc, sdst, Nn);
      // segment softmax over dst
      fill_u32_kernel<<<cdiv(Nn, 256), 256, 0, stream>>>(mbuf, 0x007FFFFFu, Nn); // ord(-inf)
      fill0(sbuf, Nn);
      fill0(agg, Nn * HDIM);
      edge_score_kernel<<<cdiv(E, 256), 256, 0, stream>>>(ssrc, sdst, srcI, dstI, ebuf, mbuf, E);
      edge_expsum_kernel<<<cdiv(E, 256), 256, 0, stream>>>(ebuf, mbuf, sbuf, dstI, E);
      {
        long long tot = (long long)E * 32;
        edge_agg_kernel<<<(unsigned)((tot + 255) / 256), 256, 0, stream>>>(
            ebuf, sbuf, zbuf, srcI, dstI, agg, E);
      }
      // BN(ReLU(agg)) -> concat column block
      fill0(bnsum, 256);
      bn_stats_kernel<<<256, HDIM, 0, stream>>>(agg, bnsum, bnsum + HDIM, Nn);
      bn_apply_kernel<<<cdiv(Nn * HDIM, 256), 256, 0, stream>>>(
          agg, bnsum, bnsum + HDIM,
          bn_gamma + (size_t)(j * HEADS + i) * HDIM, bn_beta + (size_t)(j * HEADS + i) * HDIM,
          concat, Nn, i);
    }
    // h = concat @ out_W[j] + out_b[j]   (WMMA, K=512)
    gemm(concat, outWh + (size_t)j * NCOL * HEADS * HDIM, h_cur,
         out_b + (size_t)j * HDIM, Nn, HEADS * HDIM, HDIM);
    // per-graph mean -> gru_x[1+j]
    fill0(gsum, N_GRAPHS * HDIM);
    graph_accum_kernel<<<cdiv(Nn * HDIM, 256), 256, 0, stream>>>(h_cur, gids, gsum, Nn, HDIM);
    graph_mean_kernel<<<cdiv(N_GRAPHS * HDIM, 256), 256, 0, stream>>>(
        gsum, gcount, gru_x + (size_t)(1 + j) * N_GRAPHS * HDIM, HDIM);
  }

  // ---- bidirectional 2-layer GRU, mean of 4 final hiddens -> d_out [256,128]
  GruAll P;
  for (int d = 0; d < 4; ++d) {
    P.d[d].Wih = (const float*)d_in[GRU0 + d * 4 + 0];
    P.d[d].Whh = (const float*)d_in[GRU0 + d * 4 + 1];
    P.d[d].bih = (const float*)d_in[GRU0 + d * 4 + 2];
    P.d[d].bhh = (const float*)d_in[GRU0 + d * 4 + 3];
  }
  gru_kernel<<<N_GRAPHS, HDIM, 0, stream>>>(gru_x, P, (float*)d_out);
}
